// WordLabelAttention_80461917323285
// MI455X (gfx1250) — compile-verified, matrix operand order checked by reading, not run on hardware
//
#include <hip/hip_runtime.h>

// Shapes from reference: N=8, KLEN=512, Q=16, S=128, E=256, H=8
#define NBATCH 8
#define KL     512
#define QS     2048   // Q*S
#define EDIM   256
#define HEADS  8

typedef _Float16 h8   __attribute__((ext_vector_type(8)));
typedef _Float16 v16h __attribute__((ext_vector_type(16)));
typedef float    v8f  __attribute__((ext_vector_type(8)));
typedef int      v4i  __attribute__((ext_vector_type(4)));

#if __has_builtin(__builtin_amdgcn_global_load_async_to_lds_b128)
// Diagnostic from probe: param0 is 'v4i __device__ *'  (addrspace(1) == global),
// so the call is (global_src_v4i*, lds_dst_v4i*, imm offset, imm cpol).
typedef __attribute__((address_space(1))) v4i glob_v4i;
typedef __attribute__((address_space(3))) v4i lds_v4i;
#define HAVE_ASYNC_LDS 1
#else
#define HAVE_ASYNC_LDS 0
#endif

// Build a 16x32 f16 WMMA fragment from a row-major [rows x ld] f16 matrix.
// Identical striping for A ([M][K]) and B stored as B^T ([N][K]):
//   lane<16 : halves 0..7 = K0..K0+7,  halves 8..15 = K0+16..K0+23 (row r0+lane)
//   lane>=16: halves 0..7 = K0+8..+15, halves 8..15 = K0+24..+31   (row r0+lane-16)
template <typename PtrT>
__device__ __forceinline__ v16h load_frag(PtrT base, int r0, int k0, int ld, int lane) {
  PtrT p = base + (size_t)(r0 + (lane & 15)) * ld + k0 + ((lane & 16) ? 8 : 0);
  union { v16h v; h8 h[2]; } u;
  u.h[0] = *(const h8*)(p);
  u.h[1] = *(const h8*)(p + 16);
  return u.v;
}

// C[M x N] = A[M x K] * B^T (B stored [N][K] row-major), f32 accumulate.
// Block = 256 threads = 8 waves stacked in M; wave tile = 32(M) x 64(N):
//   2 A-frags (global) x 4 B-frags (LDS, staged once per block) -> 8 WMMA / k-step.
// OutT: _Float16 or float. TRANSB: write out as [row/mb][N][mb] (batch-transposed).
template <typename OutT, bool TRANSB>
__global__ __launch_bounds__(256) void gemm_wmma_f16(
    const _Float16* __restrict__ A, long sA,
    const _Float16* __restrict__ B, long sB,
    OutT* __restrict__ C, long sC,
    int M, int N, int K, int lda, int ldb, int ldo,
    int mb, const float* __restrict__ bias) {
  extern __shared__ _Float16 bs[];  // [64][K] strip of B^T for this block's n0
  const int lane = threadIdx.x & 31;
  const int wave = threadIdx.x >> 5;
  const int n0 = blockIdx.x * 64;
  const int m0 = (blockIdx.y * 8 + wave) * 32;
  const int bz = blockIdx.z;
  A += (size_t)bz * sA;
  B += (size_t)bz * sB;
  C += (size_t)bz * sC;

  // ---- stage B[n0..n0+63][0..K) into LDS, cooperatively (async path if available)
  {
    const int kchunks = K >> 3;  // 16-byte chunks per row
    for (int c = threadIdx.x; c < 64 * kchunks; c += 256) {
      const int r = c / kchunks;
      const int kk = (c - r * kchunks) << 3;
      const _Float16* g = B + (size_t)(n0 + r) * ldb + kk;
      _Float16* l = bs + r * K + kk;
#if HAVE_ASYNC_LDS
      __builtin_amdgcn_global_load_async_to_lds_b128((glob_v4i*)g, (lds_v4i*)l, 0, 0);
#else
      *(h8*)l = *(const h8*)g;
#endif
    }
#if HAVE_ASYNC_LDS
#if __has_builtin(__builtin_amdgcn_s_wait_asynccnt)
    __builtin_amdgcn_s_wait_asynccnt(0);
#else
    asm volatile("s_wait_asynccnt 0x0" ::: "memory");
#endif
#endif
    __syncthreads();
  }

  v8f acc[2][4] = {};
  for (int k0 = 0; k0 < K; k0 += 32) {
    const v16h a0 = load_frag(A, m0, k0, lda, lane);
    const v16h a1 = load_frag(A, m0 + 16, k0, lda, lane);
#pragma unroll
    for (int t = 0; t < 4; ++t) {
      const v16h b = load_frag((const _Float16*)bs, t * 16, k0, K, lane);
      acc[0][t] =
          __builtin_amdgcn_wmma_f32_16x16x32_f16(false, a0, false, b, (short)0, acc[0][t], false, false);
      acc[1][t] =
          __builtin_amdgcn_wmma_f32_16x16x32_f16(false, a1, false, b, (short)0, acc[1][t], false, false);
    }
  }

  // D layout: VGPR j holds (M = (lane<16 ? j : 8+j), N = n-base + (lane&15))
  const int lm = lane & 15;
  const int rb = (lane >> 4) * 8;
#pragma unroll
  for (int u = 0; u < 2; ++u) {
#pragma unroll
    for (int t = 0; t < 4; ++t) {
      const int col = n0 + t * 16 + lm;
      const float badd = bias ? bias[col] : 0.0f;
#pragma unroll
      for (int j = 0; j < 8; ++j) {
        const int row = m0 + u * 16 + rb + j;
        const float v = acc[u][t][j] + badd;
        if (TRANSB) {
          C[(size_t)(row / mb) * N * mb + (size_t)col * mb + (row % mb)] = (OutT)v;
        } else {
          C[(size_t)row * ldo + col] = (OutT)v;
        }
      }
    }
  }
}

// Masked, scaled softmax over KLEN=512, one wave32 per row, in-place on f16.
__global__ __launch_bounds__(256) void softmax_mask(_Float16* __restrict__ eng,
                                                    const int* __restrict__ mask,
                                                    int rows) {
  const int lane = threadIdx.x & 31;
  const int wave = threadIdx.x >> 5;
  const int row = blockIdx.x * 8 + wave;
  if (row >= rows) return;
  const int n = row >> 11;  // row / QS
  _Float16* p = eng + (size_t)row * KL;
  const int* m = mask + n * KL;

  float x[16];
  float mx = -3.0e38f;
#pragma unroll
  for (int j = 0; j < 16; ++j) {
    const int k = j * 32 + lane;
    float e = (float)p[k] * 0.0625f;  // 1/sqrt(E) = 1/16
    if (m[k] == 0) e = -3.0e38f;      // masked_fill(-1e20) equivalent
    x[j] = e;
    mx = fmaxf(mx, e);
  }
#pragma unroll
  for (int o = 16; o > 0; o >>= 1) mx = fmaxf(mx, __shfl_xor(mx, o, 32));
  float s = 0.0f;
#pragma unroll
  for (int j = 0; j < 16; ++j) { x[j] = __expf(x[j] - mx); s += x[j]; }
#pragma unroll
  for (int o = 16; o > 0; o >>= 1) s += __shfl_xor(s, o, 32);
  const float inv = 1.0f / s;
#pragma unroll
  for (int j = 0; j < 16; ++j) p[j * 32 + lane] = (_Float16)(x[j] * inv);
}

__global__ void cvt_f32_f16(const float* __restrict__ s, _Float16* __restrict__ d, int n) {
  const int i = blockIdx.x * blockDim.x + threadIdx.x;
  if (i < n) d[i] = (_Float16)s[i];
}

// WoSum[e][d] = sum_h Wo[e][h*E + d]  (heads are replicated -> fold Wo to 256x256)
__global__ void wo_reduce(const float* __restrict__ Wo, _Float16* __restrict__ d) {
  const int i = blockIdx.x * blockDim.x + threadIdx.x;
  if (i >= EDIM * EDIM) return;
  const int e = i >> 8, dd = i & 255;
  float s = 0.0f;
#pragma unroll
  for (int h = 0; h < HEADS; ++h) s += Wo[(size_t)e * (HEADS * EDIM) + h * EDIM + dd];
  d[i] = (_Float16)s;
}

extern "C" void kernel_launch(void* const* d_in, const int* in_sizes, int n_in,
                              void* d_out, int out_size, void* d_ws, size_t ws_size,
                              hipStream_t stream) {
  const float* values = (const float*)d_in[0];
  const float* keys   = (const float*)d_in[1];
  const float* query  = (const float*)d_in[2];
  const int*   mask   = (const int*)d_in[3];
  const float* Wv = (const float*)d_in[4];
  const float* Wk = (const float*)d_in[5];
  const float* Wq = (const float*)d_in[6];
  const float* Wo = (const float*)d_in[7];
  const float* bo = (const float*)d_in[8];
  float* out = (float*)d_out;

  // Workspace carve (all f16 intermediates; ~51 MB total, L2-resident on MI455X)
  char* ws = (char*)d_ws;
  size_t off = 0;
  auto carve = [&](size_t bytes) -> void* {
    void* p = ws + off;
    off = (off + bytes + 255) & ~(size_t)255;
    return p;
  };
  _Float16* q16  = (_Float16*)carve((size_t)NBATCH * QS * EDIM * 2);  // query in f16
  _Float16* k16  = (_Float16*)carve((size_t)NBATCH * KL * EDIM * 2);  // keys  in f16
  _Float16* v16  = (_Float16*)carve((size_t)NBATCH * KL * EDIM * 2);  // values in f16
  _Float16* wq16 = (_Float16*)carve((size_t)EDIM * EDIM * 2);
  _Float16* wk16 = (_Float16*)carve((size_t)EDIM * EDIM * 2);
  _Float16* wv16 = (_Float16*)carve((size_t)EDIM * EDIM * 2);
  _Float16* wo16 = (_Float16*)carve((size_t)EDIM * EDIM * 2);         // head-folded Wo
  _Float16* qh   = (_Float16*)carve((size_t)NBATCH * QS * EDIM * 2);  // q proj
  _Float16* kh   = (_Float16*)carve((size_t)NBATCH * KL * EDIM * 2);  // k proj [n][k][d]
  _Float16* vT   = (_Float16*)carve((size_t)NBATCH * KL * EDIM * 2);  // v proj [n][d][k]
  _Float16* eng  = (_Float16*)carve((size_t)NBATCH * QS * KL * 2);    // energy/attn
  _Float16* ctx  = (_Float16*)carve((size_t)NBATCH * QS * EDIM * 2);  // attn @ v
  (void)ws_size; (void)in_sizes; (void)n_in; (void)out_size;

  auto cdiv = [](int a, int b) { return (a + b - 1) / b; };
  const dim3 blk(256);
  const size_t lds256 = 64 * 256 * sizeof(_Float16);  // B strip for K=256 (32 KB)
  const size_t lds512 = 64 * 512 * sizeof(_Float16);  // B strip for K=512 (64 KB)

  // --- precision conversion + weight folding ---
  cvt_f32_f16<<<cdiv(NBATCH * QS * EDIM, 256), blk, 0, stream>>>(query, q16, NBATCH * QS * EDIM);
  cvt_f32_f16<<<cdiv(NBATCH * KL * EDIM, 256), blk, 0, stream>>>(keys, k16, NBATCH * KL * EDIM);
  cvt_f32_f16<<<cdiv(NBATCH * KL * EDIM, 256), blk, 0, stream>>>(values, v16, NBATCH * KL * EDIM);
  cvt_f32_f16<<<cdiv(EDIM * EDIM, 256), blk, 0, stream>>>(Wq, wq16, EDIM * EDIM);
  cvt_f32_f16<<<cdiv(EDIM * EDIM, 256), blk, 0, stream>>>(Wk, wk16, EDIM * EDIM);
  cvt_f32_f16<<<cdiv(EDIM * EDIM, 256), blk, 0, stream>>>(Wv, wv16, EDIM * EDIM);
  wo_reduce<<<cdiv(EDIM * EDIM, 256), blk, 0, stream>>>(Wo, wo16);

  // --- projections: X @ W^T  (W row-major [out][in] == B^T layout directly) ---
  gemm_wmma_f16<_Float16, false><<<dim3(EDIM / 64, (NBATCH * QS) / 256, 1), blk, lds256, stream>>>(
      q16, 0, wq16, 0, qh, 0, NBATCH * QS, EDIM, EDIM, EDIM, EDIM, EDIM, 1, nullptr);
  gemm_wmma_f16<_Float16, false><<<dim3(EDIM / 64, (NBATCH * KL) / 256, 1), blk, lds256, stream>>>(
      k16, 0, wk16, 0, kh, 0, NBATCH * KL, EDIM, EDIM, EDIM, EDIM, EDIM, 1, nullptr);
  // v projection, stored transposed per batch: vT[n][d][k]
  gemm_wmma_f16<_Float16, true><<<dim3(EDIM / 64, (NBATCH * KL) / 256, 1), blk, lds256, stream>>>(
      v16, 0, wv16, 0, vT, 0, NBATCH * KL, EDIM, EDIM, EDIM, EDIM, 0, KL, nullptr);

  // --- energy[n][qs][k] = q[n][qs][:] . k[n][k][:]   (B^T = kh rows, contiguous) ---
  gemm_wmma_f16<_Float16, false><<<dim3(KL / 64, QS / 256, NBATCH), blk, lds256, stream>>>(
      qh, (long)QS * EDIM, kh, (long)KL * EDIM, eng, (long)QS * KL,
      QS, KL, EDIM, EDIM, EDIM, KL, 1, nullptr);

  // --- masked softmax over k (scale 1/16), in place ---
  softmax_mask<<<(NBATCH * QS) / 8, blk, 0, stream>>>(eng, mask, NBATCH * QS);

  // --- ctx[n][qs][d] = attn[n][qs][:] @ v[n][:][d]   (B^T = vT rows, contiguous) ---
  gemm_wmma_f16<_Float16, false><<<dim3(EDIM / 64, QS / 256, NBATCH), blk, lds512, stream>>>(
      eng, (long)QS * KL, vT, (long)EDIM * KL, ctx, (long)QS * EDIM,
      QS, EDIM, KL, KL, KL, EDIM, 1, nullptr);

  // --- out = ctx @ WoSum^T + bo  (f32 output straight to d_out) ---
  gemm_wmma_f16<float, false><<<dim3(EDIM / 64, (NBATCH * QS) / 256, 1), blk, lds256, stream>>>(
      ctx, 0, wo16, 0, out, 0, NBATCH * QS, EDIM, EDIM, EDIM, EDIM, EDIM, 1, bo);
}
